// O3onO2TensorProductWeighted_50440095924880
// MI455X (gfx1250) — matrix operand order repaired and checked by reading, not compiled, not run on hardware
//
#include <hip/hip_runtime.h>

// MI455X / gfx1250 implementation of the O3 tensor product.
// Memory-bound (512 MB HBM traffic, ~23.6 GFLOP -> ~22us HBM floor @23.3TB/s):
// f16 WMMA (16x16x32, f32 acc) keeps compute ~5x under the floor; weights stay
// hot in L2 and are held as per-wave resident B-fragments in VGPRs.
// Staging uses CDNA5 async global->LDS copies (ASYNCcnt) with LDS double
// buffering to hide HBM latency behind WMMA compute.

typedef __attribute__((ext_vector_type(16))) _Float16 v16h;
typedef __attribute__((ext_vector_type(8)))  float    v8f;
typedef __attribute__((ext_vector_type(4)))  int      v4i;

#define DIMX        1024
#define LDS_STRIDE  1028          // 1024 + 4 floats pad: breaks 4096B bank stride, keeps 16B align
#define SUBROWS     16
#define ROWS_PER_WG 128
#define NSUB        (ROWS_PER_WG / SUBROWS)
#define BUF_FLOATS  (SUBROWS * LDS_STRIDE)

#if defined(__AMDGCN__) && __has_builtin(__builtin_amdgcn_global_load_async_to_lds_b128)
#define USE_ASYNC 1
#else
#define USE_ASYNC 0
#endif

template<int N>
__device__ __forceinline__ void wait_asynccnt() {
#if defined(__AMDGCN__)
# if __has_builtin(__builtin_amdgcn_s_wait_asynccnt)
  __builtin_amdgcn_s_wait_asynccnt((short)N);
# else
  asm volatile("s_wait_asynccnt %0" :: "i"(N) : "memory");
# endif
#endif
}

__device__ __forceinline__ void async_copy_b128(const float* g, float* l) {
#if USE_ASYNC
  // Builtin params are v4i* (16B-typed pointers): (global src, LDS dst, imm offset, imm cpol)
  __builtin_amdgcn_global_load_async_to_lds_b128((v4i*)g, (v4i*)l, 0, 0);
#else
  (void)g; (void)l;
#endif
}

// Each wave stages one 1024-float row: 8 x (32 lanes x 16B) async transfers.
__device__ __forceinline__ void stage_async(const float* __restrict__ gx,
                                            float* lbuf, int wave, int lane) {
  const float* g = gx + (size_t)wave * DIMX + lane * 4;
  float*       l = lbuf + wave * LDS_STRIDE + lane * 4;
  #pragma unroll
  for (int j = 0; j < 8; ++j) async_copy_b128(g + 128 * j, l + 128 * j);
}

__device__ __forceinline__ v8f wmma_f16(v16h a, v16h b, v8f c) {
  // (neg_a, A, neg_b, B, c_mod, C, reuse_a, reuse_b)
  return __builtin_amdgcn_wmma_f32_16x16x32_f16(false, a, false, b, (short)0, c,
                                                false, false);
}

// Block offsets (floats) of each irrep block inside a row: l=0,1,2,3
template<int L> struct BlkOff { static constexpr int v = (L==0)?0:(L==1)?64:(L==2)?256:576; };

// One input block (li) contribution into the output accumulators of an (oi=LO) wave.
// A-fragment layout (16-bit A 16x32): lane&15 = row M; e<8 -> K = 8*hi + e (+32*frag),
// e>=8 -> K = 16 + 8*hi + (e-8) (+32*frag).
template<int LO, int LI>
__device__ __forceinline__ void do_block(const float* __restrict__ sx,
                                         const v16h (&bf)[2],
                                         float hz, const float (&hp)[3], const float (&hn)[3],
                                         v8f (&acc)[2*LO+1], int n, int hi)
{
  constexpr int S = 2*LI + 1;
  const float* base = sx + n * LDS_STRIDE + BlkOff<LI>::v;
  const int hoff = hi * 8;

  #pragma unroll
  for (int f = 0; f < 2; ++f) {
    // m = 0 term (hzero folded into A)
    v16h a0;
    #pragma unroll
    for (int e = 0; e < 16; ++e) {
      const int k = ((e < 8) ? 0 : 16) + hoff + (e & 7) + 32 * f;
      a0[e] = (_Float16)(hz * base[k * S + LI]);
    }
    acc[LO] = wmma_f16(a0, bf[f], acc[LO]);

    #pragma unroll
    for (int mm = 1; mm <= ((LO < LI) ? LO : LI); ++mm) {
      const float cp = hp[mm-1], cn = hn[mm-1];
      v16h aP, aM;
      #pragma unroll
      for (int e = 0; e < 16; ++e) {
        const int k = ((e < 8) ? 0 : 16) + hoff + (e & 7) + 32 * f;
        const float xp = base[k * S + LI + mm];
        const float xn = base[k * S + LI - mm];
        aP[e] = (_Float16)(cp * xp + cn * xn);   // -> component lo+mm
        aM[e] = (_Float16)(cp * xn - cn * xp);   // -> component lo-mm
      }
      acc[LO+mm] = wmma_f16(aP, bf[f], acc[LO+mm]);
      acc[LO-mm] = wmma_f16(aM, bf[f], acc[LO-mm]);
    }
  }
}

template<int LO>
__device__ __forceinline__ void compute_sub(const float* __restrict__ sx,
    const v16h (&bfrag)[4][2], const float (&hzv)[4],
    const float (&hpv)[4][3], const float (&hnv)[4][3],
    float* __restrict__ out, size_t rowbase, int o, int n, int hi)
{
  v8f acc[2*LO+1] = {};
  do_block<LO,0>(sx, bfrag[0], hzv[0], hpv[0], hnv[0], acc, n, hi);
  do_block<LO,1>(sx, bfrag[1], hzv[1], hpv[1], hnv[1], acc, n, hi);
  do_block<LO,2>(sx, bfrag[2], hzv[2], hpv[2], hnv[2], acc, n, hi);
  do_block<LO,3>(sx, bfrag[3], hzv[3], hpv[3], hnv[3], acc, n, hi);

  // C/D layout: VGPR r -> row r (lanes 0-15) / row r+8 (lanes 16-31); N col = lane&15.
  // The 2*LO+1 components are contiguous in memory per (row, channel).
  #pragma unroll
  for (int r = 0; r < 8; ++r) {
    const size_t off = (rowbase + (size_t)(r + hi*8)) * DIMX
                     + BlkOff<LO>::v + (size_t)o * (2*LO+1);
    #pragma unroll
    for (int c = 0; c < 2*LO+1; ++c) out[off + c] = acc[c][r];
  }
}

__global__ void __launch_bounds__(512, 1)
tp_o3_kernel(const float* __restrict__ x, const float* __restrict__ w,
             const float* __restrict__ hzero, const float* __restrict__ hpos,
             const float* __restrict__ hneg, float* __restrict__ out)
{
  extern __shared__ float sx[];  // 2 * BUF_FLOATS floats (double buffer)

  const int tid  = threadIdx.x;
  const int wave = tid >> 5;         // 0..15 (wave32)
  const int lane = tid & 31;
  const int oi   = wave >> 2;        // output irrep block
  const int og   = wave & 3;         // 16-wide output channel group
  const int n    = lane & 15;
  const int hi   = lane >> 4;
  const int o    = og * 16 + n;      // this lane's output channel (B/D column)

  // ---- One-time per-wave setup: resident B fragments (W^T, f16) + h coeffs ----
  // B layout (16-bit, 16x16x32): lane&15 = N col; e<8 -> K = 8*hi + e,
  // e>=8 -> K = 16 + 8*hi + (e-8) (+32*frag), matching the A fragment order.
  v16h bfrag[4][2];
  float hzv[4]; float hpv[4][3]; float hnv[4][3];
  #pragma unroll
  for (int ii = 0; ii < 4; ++ii) {
    const int p = oi * 4 + ii;
    const float* wp = w + (size_t)p * 4096 + (size_t)o * 64;  // W[o, k], k contiguous
    #pragma unroll
    for (int f = 0; f < 2; ++f) {
      v16h bb;
      #pragma unroll
      for (int e = 0; e < 16; ++e) bb[e] = (_Float16)wp[((e < 8) ? 0 : 16) + hi * 8 + (e & 7) + f * 32];
      bfrag[ii][f] = bb;
    }
    hzv[ii] = hzero[p];
    #pragma unroll
    for (int m = 0; m < 3; ++m) { hpv[ii][m] = hpos[p*3 + m]; hnv[ii][m] = hneg[p*3 + m]; }
  }

  const size_t row0 = (size_t)blockIdx.x * ROWS_PER_WG;

#if USE_ASYNC
  // Prologue: kick off subtile 0 (ASYNCcnt += 8 per wave).
  stage_async(x + row0 * DIMX, sx, wave, lane);
#endif

  for (int sub = 0; sub < NSUB; ++sub) {
    const size_t rowbase = row0 + (size_t)sub * SUBROWS;
    float* cur = sx + (sub & 1) * BUF_FLOATS;        // no LDS-pointer array:
    float* nxt = sx + ((sub + 1) & 1) * BUF_FLOATS;  // keep addr math runtime-only

    __syncthreads();  // buffer `nxt` no longer being read by any wave

#if USE_ASYNC
    // Issue next subtile into the other buffer, then wait for current one.
    if (sub + 1 < NSUB) {
      stage_async(x + (rowbase + SUBROWS) * DIMX, nxt, wave, lane);
      wait_asynccnt<8>();   // async loads complete in order: first 8 (cur) done
    } else {
      wait_asynccnt<0>();
    }
#else
    // Fallback: synchronous stage, wave r copies row r as 8x float4 per lane.
    {
      const float4* src = (const float4*)(x + (rowbase + wave) * DIMX);
      float4*       dst = (float4*)(cur + wave * LDS_STRIDE);
      #pragma unroll
      for (int j = 0; j < 8; ++j) dst[lane + 32*j] = src[lane + 32*j];
    }
    if (sub + 1 < NSUB)
      __builtin_prefetch(x + (rowbase + SUBROWS + wave) * DIMX + lane * 32, 0, 1);
    (void)nxt;
#endif

    __syncthreads();  // all waves' rows landed in cur

    switch (oi) {
      case 0: compute_sub<0>(cur, bfrag, hzv, hpv, hnv, out, rowbase, o, n, hi); break;
      case 1: compute_sub<1>(cur, bfrag, hzv, hpv, hnv, out, rowbase, o, n, hi); break;
      case 2: compute_sub<2>(cur, bfrag, hzv, hpv, hnv, out, rowbase, o, n, hi); break;
      default: compute_sub<3>(cur, bfrag, hzv, hpv, hnv, out, rowbase, o, n, hi); break;
    }
  }
}

extern "C" void kernel_launch(void* const* d_in, const int* in_sizes, int n_in,
                              void* d_out, int out_size, void* d_ws, size_t ws_size,
                              hipStream_t stream) {
  const float* x  = (const float*)d_in[0];
  const float* w  = (const float*)d_in[1];
  const float* hz = (const float*)d_in[2];
  const float* hp = (const float*)d_in[3];
  const float* hn = (const float*)d_in[4];
  float* out = (float*)d_out;

  const int nrows  = in_sizes[0] / DIMX;          // 65536
  const int blocks = nrows / ROWS_PER_WG;         // 512
  const size_t shmem = (size_t)2 * BUF_FLOATS * sizeof(float);  // ~128.5 KB

  hipLaunchKernelGGL(tp_o3_kernel, dim3(blocks), dim3(512), shmem, stream,
                     x, w, hz, hp, hn, out);
}
